// GlobalSparseAttention_50938312130950
// MI455X (gfx1250) — matrix operand order repaired and checked by reading, not compiled, not run on hardware
//
#include <hip/hip_runtime.h>
#include <hip/hip_bf16.h>

// ---------------------------------------------------------------------------
// Problem constants (match reference): B=1, N=2048, C=1024, H=16, D=64, S=8
// ---------------------------------------------------------------------------
#define TN   2048      // tokens
#define TC   1024      // channels
#define TH   16        // heads
#define TD   64        // head dim
#define TS   8         // frames
#define TPF  256       // tokens per frame
#define C3   3072      // 3*C

typedef _Float16 h4  __attribute__((ext_vector_type(4)));
typedef _Float16 h8  __attribute__((ext_vector_type(8)));
typedef _Float16 h16 __attribute__((ext_vector_type(16)));
typedef float    f8  __attribute__((ext_vector_type(8)));

static __device__ inline h16 hcat(h8 lo, h8 hi) {
  return __builtin_shufflevector(lo, hi, 0,1,2,3,4,5,6,7,8,9,10,11,12,13,14,15);
}

static __device__ inline f8 wmma_f16(h16 a, h16 b, f8 c) {
  // D = A(16x32 f16) * B(32x16 f16) + C(16x16 f32)
  return __builtin_amdgcn_wmma_f32_16x16x32_f16(false, a, false, b, (short)0, c,
                                                false, false);
}

// Per-lane async 16-byte copy global -> LDS (ASYNCcnt-tracked, CDNA5).
// lds_off: raw LDS byte offset (low 32 bits of the generic shared address).
static __device__ inline void async_copy16(unsigned lds_off, const void* g) {
  asm volatile("global_load_async_to_lds_b128 %0, %1, off"
               :: "v"(lds_off), "v"(g) : "memory");
}
static __device__ inline unsigned lds_addr(const void* p) {
  return (unsigned)(unsigned long long)p;   // low 32 bits = LDS offset
}

// A fragment (16x32 f16) from row-major [row][k] storage, leading dim ld.
// Lane layout (contiguous k order used by this code): half h=lane>>4, row r=lane&15:
//   V0..7 = K(16h + 0..15)  -> two 16B loads.
static __device__ inline h16 load_a_rm(const _Float16* base, int ld, int row0,
                                       int k0, int lane) {
  int r  = row0 + (lane & 15);
  int hh = lane >> 4;
  const _Float16* p = base + (long)r * ld + k0 + 16 * hh;
  h8 lo = *(const h8*)p;
  h8 hi = *(const h8*)(p + 8);
  return hcat(lo, hi);
}

// B fragment (32x16 f16) from K-contiguous storage: element (kk, col) lives at
// base + (c0+col)*ld + k0 + kk.  Lane layout: col=lane&15, half h=lane>>4,
// V0..7 = K(16h + 0..15) -> one contiguous 32B read (two 16B loads).
static __device__ inline h16 load_b_kc(const _Float16* base, int ld, int k0,
                                       int c0, int lane) {
  const _Float16* p = base + (long)(c0 + (lane & 15)) * ld + k0 + 16 * (lane >> 4);
  h8 lo = *(const h8*)p;
  h8 hi = *(const h8*)(p + 8);
  return hcat(lo, hi);
}

// ---------------------------------------------------------------------------
// Kernel 1: fp32 -> f16 convert (x), vectorized x4
// ---------------------------------------------------------------------------
__global__ void cvt_f32_f16(const float* __restrict__ src,
                            _Float16* __restrict__ dst, int n4) {
  int i = blockIdx.x * blockDim.x + threadIdx.x;
  if (i < n4) {
    float4 v = ((const float4*)src)[i];
    h4 o;
    o.x = (_Float16)v.x; o.y = (_Float16)v.y;
    o.z = (_Float16)v.z; o.w = (_Float16)v.w;
    ((h4*)dst)[i] = o;
  }
}

// ---------------------------------------------------------------------------
// Kernel 2: transpose + convert weights: src rows x cols fp32 -> dst cols x rows f16
// (makes the GEMM B operand K-contiguous so a B fragment is one 32B read)
// ---------------------------------------------------------------------------
__global__ __launch_bounds__(256) void transpose_cvt(
    const float* __restrict__ src, _Float16* __restrict__ dst, int rows, int cols) {
  __shared__ float tile[32][33];
  int c0 = blockIdx.x * 32, r0 = blockIdx.y * 32;
  int tx = threadIdx.x, ty = threadIdx.y;
#pragma unroll
  for (int i = 0; i < 32; i += 8)
    tile[ty + i][tx] = src[(long)(r0 + ty + i) * cols + c0 + tx];
  __syncthreads();
#pragma unroll
  for (int i = 0; i < 32; i += 8)
    dst[(long)(c0 + ty + i) * rows + r0 + tx] = (_Float16)tile[tx][ty + i];
}

// ---------------------------------------------------------------------------
// Kernel 3: QKV GEMM.  X(2048x1024 f16) @ WqkvT(3072x1024 f16, K-contig) + bqkv
// Block: 256 threads (8 waves), tile 128 rows x 64 cols, K-step 32.
// B tile double-buffered in LDS via async global->LDS copies.
// Scatters into Q[h][n][d], K[h][n][d], VT[h][d][n] (all f16).
// ---------------------------------------------------------------------------
__global__ __launch_bounds__(256) void qkv_gemm(
    const _Float16* __restrict__ Xh, const _Float16* __restrict__ WT,
    const float* __restrict__ bqkv, _Float16* __restrict__ Qb,
    _Float16* __restrict__ Kb, _Float16* __restrict__ VTb) {
  __shared__ __align__(16) _Float16 Bt[2][64][32];   // 2 x 4KB, K-contiguous
  int colbase = blockIdx.x * 64;
  int rowbase = blockIdx.y * 128;
  int lane = threadIdx.x & 31, wave = threadIdx.x >> 5;
  int wrow = rowbase + wave * 16;
  int hh = lane >> 4, cc = lane & 15;

  // per-thread 16B async-copy slot: col bc (0..63), 16B chunk bp (0..3)
  int bc = threadIdx.x >> 2, bp = threadIdx.x & 3;
  const _Float16* bsrc = WT + (long)(colbase + bc) * TC + bp * 8;
  unsigned lds0 = lds_addr(&Bt[0][0][0]) + (unsigned)((bc * 32 + bp * 8) * 2);

  f8 acc[4] = {};
  async_copy16(lds0, bsrc);                       // prefetch kb=0 into buf0
  int buf = 0;
  for (int kb = 0; kb < TC; kb += 32, buf ^= 1) {
    if (kb + 32 < TC) {
      async_copy16(lds0 + (unsigned)((buf ^ 1) * 4096), bsrc + kb + 32);
      asm volatile("s_wait_asynccnt 0x1" ::: "memory");
    } else {
      asm volatile("s_wait_asynccnt 0x0" ::: "memory");
    }
    __syncthreads();
    h16 a = load_a_rm(Xh, TC, wrow, kb, lane);
    const _Float16* bbase = &Bt[buf][0][0];
#pragma unroll
    for (int t = 0; t < 4; t++) {
      h16 b = load_b_kc(bbase, 32, 0, t * 16, lane);
      acc[t] = wmma_f16(a, b, acc[t]);
    }
    __syncthreads();
  }

  // scatter: col = which*1024 + head*64 + d
#pragma unroll
  for (int t = 0; t < 4; t++) {
    int col = colbase + t * 16 + cc;
    float bias = bqkv[col];
    int which = col >> 10;
    int rem = col & 1023;
    int head = rem >> 6;
    int d = rem & 63;
    if (which == 2) {           // VT[h][d][n]: contiguous over token -> h8 store
      h8 vv;
#pragma unroll
      for (int j = 0; j < 8; j++) vv[j] = (_Float16)(acc[t][j] + bias);
      *(h8*)(VTb + (long)head * TD * TN + (long)d * TN + wrow + hh * 8) = vv;
    } else {
      _Float16* dstb = (which == 0 ? Qb : Kb) + (long)head * TN * TD + d;
#pragma unroll
      for (int j = 0; j < 8; j++) {
        int token = wrow + hh * 8 + j;
        dstb[(long)token * TD] = (_Float16)(acc[t][j] + bias);
      }
    }
  }
}

// ---------------------------------------------------------------------------
// Kernel 4: frame-sparse flash attention (one head per blockIdx.y).
// Block: 256 threads (8 waves); wave owns 16 query rows; 128 q-rows per block.
// K/V tiles async-staged to LDS once per 32-key block (8x dedup across waves);
// whole key frames skipped when adj[fq][kf]==0 (block-uniform).
// ---------------------------------------------------------------------------
__global__ __launch_bounds__(256) void attn_kernel(
    const _Float16* __restrict__ Qb, const _Float16* __restrict__ Kb,
    const _Float16* __restrict__ VTb, const int* __restrict__ frame_ids,
    const unsigned char* __restrict__ is_hub, const unsigned char* __restrict__ adj,
    const float* __restrict__ frame_bias, _Float16* __restrict__ AOut) {
  __shared__ __align__(16) _Float16 Kt[32][64];      // 4KB: [key][d]
  __shared__ __align__(16) _Float16 Vt[64][32];      // 4KB: [d][key]
  __shared__ __align__(16) _Float16 Plds[8][16][32]; // 8KB: per-wave P relayout
  int h = blockIdx.y;
  int qtile = blockIdx.x;                            // 16 tiles of 128 q-rows
  int lane = threadIdx.x & 31, wave = threadIdx.x >> 5;
  int qrow0 = qtile * 128 + wave * 16;
  int fq = frame_ids[qtile * 128];                   // uniform across block
  int hh = lane >> 4, cc = lane & 15;

  const _Float16* Qh = Qb + (long)h * TN * TD;
  const _Float16* Kh = Kb + (long)h * TN * TD;
  const _Float16* Vh = VTb + (long)h * TD * TN;

  // per-thread async-copy slots for the K (32x64) and V (64x32) tiles
  int kr = threadIdx.x >> 3, kch = threadIdx.x & 7;  // K: row 0..31, chunk 0..7
  int vd = threadIdx.x >> 2, vch = threadIdx.x & 3;  // V: row 0..63, chunk 0..3
  unsigned klds = lds_addr(&Kt[0][0]) + (unsigned)((kr * 64 + kch * 8) * 2);
  unsigned vlds = lds_addr(&Vt[0][0]) + (unsigned)((vd * 32 + vch * 8) * 2);

  h16 aq0 = load_a_rm(Qh, TD, qrow0, 0, lane);
  h16 aq1 = load_a_rm(Qh, TD, qrow0, 32, lane);

  float hubq[8];
#pragma unroll
  for (int j = 0; j < 8; j++) hubq[j] = is_hub[qrow0 + hh * 8 + j] ? 1.f : 0.f;

  float m[8], l[8];
  f8 O[4] = {};
#pragma unroll
  for (int j = 0; j < 8; j++) { m[j] = -3.0e38f; l[j] = 0.f; }

  for (int kf = 0; kf < TS; kf++) {
    if (!adj[fq * TS + kf]) continue;                // frame-level sparsity skip
    float fbias = frame_bias[fq * TS + kf];
    bool same = (kf == fq);

    for (int kb = kf * TPF; kb < kf * TPF + TPF; kb += 32) {
      // ---- async-stage K and V tiles for this 32-key block
      async_copy16(klds, Kh + (long)(kb + kr) * TD + kch * 8);
      async_copy16(vlds, Vh + (long)vd * TN + kb + vch * 8);
      asm volatile("s_wait_asynccnt 0x0" ::: "memory");
      __syncthreads();

      // ---- scores: two 16x16 tiles, K-dim = d = 64 (2 chained WMMAs each)
      f8 s0 = {}, s1 = {};
      {
        h16 b00 = load_b_kc(&Kt[0][0], TD, 0,  0,  lane);
        h16 b01 = load_b_kc(&Kt[0][0], TD, 32, 0,  lane);
        s0 = wmma_f16(aq0, b00, s0);
        s0 = wmma_f16(aq1, b01, s0);
        h16 b10 = load_b_kc(&Kt[0][0], TD, 0,  16, lane);
        h16 b11 = load_b_kc(&Kt[0][0], TD, 32, 16, lane);
        s1 = wmma_f16(aq0, b10, s1);
        s1 = wmma_f16(aq1, b11, s1);
      }

      float hubk0 = same ? 0.f : (is_hub[kb + cc] ? 1.f : 0.f);
      float hubk1 = same ? 0.f : (is_hub[kb + 16 + cc] ? 1.f : 0.f);

      float mt[8];
#pragma unroll
      for (int j = 0; j < 8; j++) {
        float x0 = s0[j] * 0.125f + fbias;           // 1/sqrt(64) then bias
        float x1 = s1[j] * 0.125f + fbias;
        bool blkq = !same && (hubq[j] != 0.f);
        if (blkq || hubk0 != 0.f) x0 = -1e30f;
        if (blkq || hubk1 != 0.f) x1 = -1e30f;
        s0[j] = x0; s1[j] = x1;
        mt[j] = fmaxf(x0, x1);
      }
      // row-wise max across the 16 lanes holding this row's columns
#pragma unroll
      for (int j = 0; j < 8; j++) {
        float v = mt[j];
        for (int msk = 1; msk < 16; msk <<= 1) v = fmaxf(v, __shfl_xor(v, msk, 32));
        mt[j] = v;
      }

      float al[8];
#pragma unroll
      for (int j = 0; j < 8; j++) {
        float mn = fmaxf(m[j], mt[j]);
        float alpha = __expf(m[j] - mn);
        float p0 = __expf(s0[j] - mn);
        float p1 = __expf(s1[j] - mn);
        s0[j] = p0; s1[j] = p1;
        float r = p0 + p1;
        for (int msk = 1; msk < 16; msk <<= 1) r += __shfl_xor(r, msk, 32);
        m[j] = mn;
        l[j] = l[j] * alpha + r;
        al[j] = alpha;
      }
#pragma unroll
      for (int t = 0; t < 4; t++)
#pragma unroll
        for (int j = 0; j < 8; j++) O[t][j] *= al[j];

      // ---- P: C-fragment layout -> A-fragment layout via per-wave LDS
#pragma unroll
      for (int j = 0; j < 8; j++) {
        Plds[wave][hh * 8 + j][cc]      = (_Float16)s0[j];
        Plds[wave][hh * 8 + j][16 + cc] = (_Float16)s1[j];
      }
      asm volatile("s_wait_dscnt 0x0" ::: "memory"); // own-wave LDS RAW
      h16 ap = load_a_rm(&Plds[wave][0][0], 32, 0, 0, lane);

      // ---- O += P(16x32) @ V(32 keys x 64 d): 4 d-tiles from LDS V tile
#pragma unroll
      for (int t = 0; t < 4; t++) {
        h16 bv = load_b_kc(&Vt[0][0], 32, 0, t * 16, lane);
        O[t] = wmma_f16(ap, bv, O[t]);
      }
      __syncthreads();   // protect K/V tiles before next block overwrites
    }
  }

  // normalize and store to [n][h*64+d] f16 (A-matrix for the projection GEMM)
#pragma unroll
  for (int j = 0; j < 8; j++) {
    float inv = 1.0f / l[j];
    int token = qrow0 + hh * 8 + j;
#pragma unroll
    for (int t = 0; t < 4; t++) {
      AOut[(long)token * TC + h * TD + t * 16 + cc] = (_Float16)(O[t][j] * inv);
    }
  }
}

// ---------------------------------------------------------------------------
// Kernel 5: output projection.  A(2048x1024 f16) @ WprojT(1024x1024 f16) + bproj
// ---------------------------------------------------------------------------
__global__ __launch_bounds__(256) void proj_gemm(
    const _Float16* __restrict__ Ah, const _Float16* __restrict__ WT,
    const float* __restrict__ bproj, float* __restrict__ out) {
  __shared__ __align__(16) _Float16 Bt[2][64][32];
  int colbase = blockIdx.x * 64;
  int rowbase = blockIdx.y * 128;
  int lane = threadIdx.x & 31, wave = threadIdx.x >> 5;
  int wrow = rowbase + wave * 16;
  int hh = lane >> 4, cc = lane & 15;

  int bc = threadIdx.x >> 2, bp = threadIdx.x & 3;
  const _Float16* bsrc = WT + (long)(colbase + bc) * TC + bp * 8;
  unsigned lds0 = lds_addr(&Bt[0][0][0]) + (unsigned)((bc * 32 + bp * 8) * 2);

  f8 acc[4] = {};
  async_copy16(lds0, bsrc);
  int buf = 0;
  for (int kb = 0; kb < TC; kb += 32, buf ^= 1) {
    if (kb + 32 < TC) {
      async_copy16(lds0 + (unsigned)((buf ^ 1) * 4096), bsrc + kb + 32);
      asm volatile("s_wait_asynccnt 0x1" ::: "memory");
    } else {
      asm volatile("s_wait_asynccnt 0x0" ::: "memory");
    }
    __syncthreads();
    h16 a = load_a_rm(Ah, TC, wrow, kb, lane);
    const _Float16* bbase = &Bt[buf][0][0];
#pragma unroll
    for (int t = 0; t < 4; t++) {
      h16 b = load_b_kc(bbase, 32, 0, t * 16, lane);
      acc[t] = wmma_f16(a, b, acc[t]);
    }
    __syncthreads();
  }
#pragma unroll
  for (int t = 0; t < 4; t++) {
    int col = colbase + t * 16 + cc;
    float bias = bproj[col];
#pragma unroll
    for (int j = 0; j < 8; j++) {
      int row = wrow + hh * 8 + j;
      out[(long)row * TC + col] = acc[t][j] + bias;
    }
  }
}

// ---------------------------------------------------------------------------
// Launch
// ---------------------------------------------------------------------------
extern "C" void kernel_launch(void* const* d_in, const int* in_sizes, int n_in,
                              void* d_out, int out_size, void* d_ws, size_t ws_size,
                              hipStream_t stream) {
  const float*         x          = (const float*)d_in[0];
  const int*           frame_ids  = (const int*)d_in[1];
  const unsigned char* is_hub     = (const unsigned char*)d_in[2];
  const unsigned char* adj        = (const unsigned char*)d_in[3];
  const float*         frame_bias = (const float*)d_in[4];
  const float*         Wqkv       = (const float*)d_in[5];
  const float*         bqkv       = (const float*)d_in[6];
  const float*         Wproj      = (const float*)d_in[7];
  const float*         bproj      = (const float*)d_in[8];
  float* out = (float*)d_out;

  // workspace partition (f16 buffers)
  char* ws = (char*)d_ws;
  _Float16* Xh     = (_Float16*)ws; ws += (size_t)TN * TC * 2;       // 4 MB
  _Float16* WqkvT  = (_Float16*)ws; ws += (size_t)C3 * TC * 2;       // 6 MB
  _Float16* WprojT = (_Float16*)ws; ws += (size_t)TC * TC * 2;       // 2 MB
  _Float16* Qb     = (_Float16*)ws; ws += (size_t)TH * TN * TD * 2;  // 4 MB
  _Float16* Kb     = (_Float16*)ws; ws += (size_t)TH * TN * TD * 2;  // 4 MB
  _Float16* VTb    = (_Float16*)ws; ws += (size_t)TH * TD * TN * 2;  // 4 MB
  _Float16* AOut   = (_Float16*)ws; ws += (size_t)TN * TC * 2;       // 4 MB

  // 1) x -> f16
  {
    int n4 = TN * TC / 4;
    cvt_f32_f16<<<(n4 + 255) / 256, 256, 0, stream>>>(x, Xh, n4);
  }
  // 2) weights -> transposed (K-contiguous) f16
  {
    dim3 blk(32, 8);
    transpose_cvt<<<dim3(C3 / 32, TC / 32), blk, 0, stream>>>(Wqkv, WqkvT, TC, C3);
    transpose_cvt<<<dim3(TC / 32, TC / 32), blk, 0, stream>>>(Wproj, WprojT, TC, TC);
  }
  // 3) QKV projection into attention-friendly layouts
  {
    dim3 grid(C3 / 64, TN / 128);
    qkv_gemm<<<grid, 256, 0, stream>>>(Xh, WqkvT, bqkv, Qb, Kb, VTb);
  }
  // 4) frame-sparse flash attention
  {
    dim3 grid(TN / 128, TH);
    attn_kernel<<<grid, 256, 0, stream>>>(Qb, Kb, VTb, frame_ids, is_hub, adj,
                                          frame_bias, AOut);
  }
  // 5) output projection
  {
    dim3 grid(TC / 64, TN / 128);
    proj_gemm<<<grid, 256, 0, stream>>>(AOut, WprojT, bproj, out);
  }
}
